// SGAT16_3496103379554
// MI455X (gfx1250) — matrix operand order repaired and checked by reading, not compiled, not run on hardware
//
#include <hip/hip_runtime.h>
#include <math.h>

// Problem constants (match reference)
#define NN   50000
#define EE   800000
#define ET   850000   /* EE + NN self loops */
#define HH   32
#define DIN  128
#define DOUT 16
#define LL   15
#define NT   (NN / 16)   /* 3125 row tiles */
#define TM   4           /* row tiles per wave (hidden/out GEMMs) */
#define TMIN 2           /* row tiles per wave (input GEMM; 8 B-frags live -> lower TM avoids spills) */

typedef __attribute__((ext_vector_type(16))) _Float16 v16h;
typedef __attribute__((ext_vector_type(8)))  float    v8f;

static __device__ __forceinline__ v8f wmma_f16(v16h a, v16h b, v8f c) {
  // v_wmma_f32_16x16x32_f16 : D = A(16x32,f16) * B(32x16,f16) + C(16x16,f32)
  return __builtin_amdgcn_wmma_f32_16x16x32_f16(
      /*neg_a=*/false, a, /*neg_b=*/false, b,
      /*c_mod=*/(short)0, c, /*reuse_a=*/false, /*reuse_b=*/false);
}

// float atomic max via monotonic int/uint bit trick (amax initialized to -inf)
static __device__ __forceinline__ void atomicMaxF(float* addr, float v) {
  if (v >= 0.f) atomicMax((int*)addr, __float_as_int(v));
  else          atomicMin((unsigned int*)addr, __float_as_uint(v));
}

// ---------------- weight pre-pack: f32 [K,ncols] -> f16 WMMA B fragments --------
// Fragment (kt, nt): 32 lanes x 16 halfs contiguous (512 halfs / 1KB).
// B layout: lane holds col n = nt*16 + (lane&15); kb = kt*32 + 16*(j>>3) + 8*(lane>>4) + (j&7).
__global__ void k_pack_w(const float* __restrict__ W, int ncols, int ktiles, int ntiles,
                         _Float16* __restrict__ dst) {
  int idx = blockIdx.x * blockDim.x + threadIdx.x;
  int total = ktiles * ntiles * 32;
  if (idx >= total) return;
  int lane = idx & 31;
  int nt = (idx >> 5) % ntiles;
  int kt = idx / (32 * ntiles);
  int n = nt * 16 + (lane & 15);
  int half = lane >> 4;
  _Float16* o = dst + ((kt * ntiles + nt) * 32 + lane) * 16;
#pragma unroll
  for (int j = 0; j < 16; ++j) {
    int kb = kt * 32 + ((j >> 3) << 4) + (half << 3) + (j & 7);   // table order, matches A
    o[j] = (_Float16)W[kb * ncols + n];
  }
}

// ---------------- GEMM kernels (WMMA f16->f32) ----------------
// A fragment: lane holds row M=lane&15, ka = 16*(v>>2) + 8*(lane>>4) + 2*(v&3).
// C/D: VGPR r, lane L -> row (L>>4)*8+r, col L&15.

// h[N,32] = x[N,128] @ W0[128,32] + b0   (B pre-packed: 4 ktiles x 2 ntiles)
__global__ void k_gemm_in(const float* __restrict__ x, const _Float16* __restrict__ wp,
                          const float* __restrict__ b0, float* __restrict__ h) {
  int wid = (blockIdx.x * blockDim.x + threadIdx.x) >> 5;
  int lane = threadIdx.x & 31;
  if (wid * TMIN >= NT) return;             // wave-uniform exit
  int m = lane & 15, half = lane >> 4;
  v16h B[8];
#pragma unroll
  for (int f = 0; f < 8; ++f) B[f] = *(const v16h*)(wp + (f * 32 + lane) * 16);
#pragma unroll 1
  for (int t = 0; t < TMIN; ++t) {
    int tile = wid * TMIN + t;
    if (tile >= NT) break;                  // wave-uniform
    int row = tile * 16 + m;
    v8f c0 = {}, c1 = {};
#pragma unroll
    for (int kt = 0; kt < 4; ++kt) {
      v16h a;
#pragma unroll
      for (int v = 0; v < 8; ++v) {
        int ka = kt * 32 + ((v >> 2) << 4) + (half << 3) + ((v & 3) << 1);
        float2 xa = *(const float2*)(x + row * DIN + ka);
        a[2 * v] = (_Float16)xa.x; a[2 * v + 1] = (_Float16)xa.y;
      }
      c0 = wmma_f16(a, B[kt * 2 + 0], c0);
      c1 = wmma_f16(a, B[kt * 2 + 1], c1);
    }
#pragma unroll
    for (int r = 0; r < 8; ++r) {
      int orow = tile * 16 + half * 8 + r;
      h[orow * HH + m]      = c0[r] + b0[m];
      h[orow * HH + m + 16] = c1[r] + b0[m + 16];
    }
  }
}

// hp[N,32] = h[N,32] @ W[32,32]   (B pre-packed: 2 ntiles)
__global__ void k_gemm_hid(const float* __restrict__ h, const _Float16* __restrict__ wp,
                           float* __restrict__ hp) {
  int wid = (blockIdx.x * blockDim.x + threadIdx.x) >> 5;
  int lane = threadIdx.x & 31;
  if (wid * TM >= NT) return;
  int m = lane & 15, half = lane >> 4;
  v16h bf0 = *(const v16h*)(wp + (0 * 32 + lane) * 16);
  v16h bf1 = *(const v16h*)(wp + (1 * 32 + lane) * 16);
  for (int t = 0; t < TM; ++t) {
    int tile = wid * TM + t;
    if (tile >= NT) break;
    int row = tile * 16 + m;
    v16h a;
#pragma unroll
    for (int v = 0; v < 8; ++v) {
      int ka = ((v >> 2) << 4) + (half << 3) + ((v & 3) << 1);
      float2 xa = *(const float2*)(h + row * HH + ka);
      a[2 * v] = (_Float16)xa.x; a[2 * v + 1] = (_Float16)xa.y;
    }
    v8f c0 = {}, c1 = {};
    c0 = wmma_f16(a, bf0, c0);
    c1 = wmma_f16(a, bf1, c1);
#pragma unroll
    for (int r = 0; r < 8; ++r) {
      int orow = tile * 16 + half * 8 + r;
      hp[orow * HH + m]      = c0[r];
      hp[orow * HH + m + 16] = c1[r];
    }
  }
}

// out[N,16] = h[N,32] @ W16[32,16] + b16   (B pre-packed: 1 fragment)
__global__ void k_gemm_out(const float* __restrict__ h, const _Float16* __restrict__ wp,
                           const float* __restrict__ b16, float* __restrict__ out) {
  int wid = (blockIdx.x * blockDim.x + threadIdx.x) >> 5;
  int lane = threadIdx.x & 31;
  if (wid * TM >= NT) return;
  int m = lane & 15, half = lane >> 4;
  v16h bf0 = *(const v16h*)(wp + lane * 16);
  for (int t = 0; t < TM; ++t) {
    int tile = wid * TM + t;
    if (tile >= NT) break;
    int row = tile * 16 + m;
    v16h a;
#pragma unroll
    for (int v = 0; v < 8; ++v) {
      int ka = ((v >> 2) << 4) + (half << 3) + ((v & 3) << 1);
      float2 xa = *(const float2*)(h + row * HH + ka);
      a[2 * v] = (_Float16)xa.x; a[2 * v + 1] = (_Float16)xa.y;
    }
    v8f c0 = {};
    c0 = wmma_f16(a, bf0, c0);
#pragma unroll
    for (int r = 0; r < 8; ++r) {
      int orow = tile * 16 + half * 8 + r;
      out[orow * DOUT + m] = c0[r] + b16[m];
    }
  }
}

// ---------------- per-node attention scores ----------------
__global__ void k_scores(const float* __restrict__ hp, const float* __restrict__ al,
                         const float* __restrict__ ar, float* __restrict__ sL,
                         float* __restrict__ sR) {
  int n = blockIdx.x * blockDim.x + threadIdx.x;
  if (n >= NN) return;
  float a = 0.f, b = 0.f;
#pragma unroll
  for (int c = 0; c < HH; ++c) {
    float v = hp[n * HH + c];
    a += v * al[c];
    b += v * ar[c];
  }
  sL[n] = a; sR[n] = b;
}

// ---------------- per-layer init: amax=-inf, den=0, acc=0 ----------------
__global__ void k_init(float* __restrict__ amax, float* __restrict__ den,
                       float* __restrict__ acc) {
  int i = blockIdx.x * blockDim.x + threadIdx.x;
  if (i < NN) { amax[i] = -INFINITY; den[i] = 0.f; }
  if (i < NN * HH) acc[i] = 0.f;
}

// ---------------- edge pass 1: alpha + segment max (wave per edge) ----------------
__global__ void k_edge1(const int* __restrict__ ei, const float* __restrict__ hp,
                        const float* __restrict__ sL, const float* __restrict__ sR,
                        float* __restrict__ alpha, float* __restrict__ amax) {
  int wave = (blockIdx.x * blockDim.x + threadIdx.x) >> 5;
  int lane = threadIdx.x & 31;
  if (wave >= ET) return;
  int s, d;
  if (wave < EE) { s = ei[wave]; d = ei[EE + wave]; } else { s = d = wave - EE; }
  // coalesced 128B gathers: lane = channel (wave32 == H exactly)
  float p = hp[s * HH + lane] * hp[d * HH + lane];
#pragma unroll
  for (int off = 16; off > 0; off >>= 1) p += __shfl_xor(p, off);
  if (lane == 0) {
    float logit = p;
    float a = sL[s] + sR[d];
    a = a / (1.f + __expf(-logit));      // MX gating
    a = (a >= 0.f) ? a : 0.2f * a;       // leaky_relu
    alpha[wave] = a;
    atomicMaxF(&amax[d], a);
  }
}

// ---------------- edge pass 2: exp + segment sum (thread per edge) ----------------
__global__ void k_edge2(const int* __restrict__ ei, float* __restrict__ alpha,
                        const float* __restrict__ amax, float* __restrict__ den) {
  int e = blockIdx.x * blockDim.x + threadIdx.x;
  if (e >= ET) return;
  int d = (e < EE) ? ei[EE + e] : (e - EE);
  float ex = __expf(alpha[e] - amax[d]);
  alpha[e] = ex;                          // overwrite with exp value
  atomicAdd(&den[d], ex);
}

// ---------------- edge pass 3: weighted scatter (wave per edge) ----------------
__global__ void k_edge3(const int* __restrict__ ei, const float* __restrict__ hp,
                        const float* __restrict__ alpha, const float* __restrict__ den,
                        float* __restrict__ acc) {
  int wave = (blockIdx.x * blockDim.x + threadIdx.x) >> 5;
  int lane = threadIdx.x & 31;
  if (wave >= ET) return;
  int s, d;
  if (wave < EE) { s = ei[wave]; d = ei[EE + wave]; } else { s = d = wave - EE; }
  float coef = alpha[wave] / den[d];
  atomicAdd(&acc[d * HH + lane], hp[s * HH + lane] * coef);
}

// ---------------- epilogue: h = relu(acc + b) ----------------
__global__ void k_epilogue(const float* __restrict__ acc, const float* __restrict__ b,
                           float* __restrict__ h) {
  int i = blockIdx.x * blockDim.x + threadIdx.x;
  if (i >= NN * HH) return;
  float v = acc[i] + b[i & (HH - 1)];
  h[i] = (v > 0.f) ? v : 0.f;
}

extern "C" void kernel_launch(void* const* d_in, const int* in_sizes, int n_in,
                              void* d_out, int out_size, void* d_ws, size_t ws_size,
                              hipStream_t stream) {
  (void)in_sizes; (void)n_in; (void)out_size; (void)ws_size;
  const float* x     = (const float*)d_in[0];
  const int*   ei    = (const int*)  d_in[1];
  const float* W0    = (const float*)d_in[2];
  const float* b0    = (const float*)d_in[3];
  const float* Ws    = (const float*)d_in[4];   // [L,32,32]
  const float* att_l = (const float*)d_in[5];   // [L,32]
  const float* att_r = (const float*)d_in[6];   // [L,32]
  const float* bs    = (const float*)d_in[7];   // [L,32]
  const float* W16   = (const float*)d_in[8];
  const float* b16   = (const float*)d_in[9];
  float* out = (float*)d_out;

  // workspace layout (floats)
  float* ws    = (float*)d_ws;
  float* h     = ws;                    // 1,600,000
  float* hp    = ws + 1600000;          // 1,600,000
  float* acc   = ws + 3200000;          // 1,600,000
  float* alpha = ws + 4800000;          //   850,000
  float* sL    = ws + 5650048;          //    50,000
  float* sR    = ws + 5700048;
  float* amax  = ws + 5750048;
  float* den   = ws + 5800048;
  // packed f16 weight fragments (32B-aligned): ~40 KB
  _Float16* pW0  = (_Float16*)(ws + 5850048);   //  4 kt x 2 nt x 512 = 4096 halfs
  _Float16* pWs  = pW0 + 4096;                  // 15 kt x 2 nt x 512 = 15360 halfs
  _Float16* pW16 = pWs + 15360;                 //  1 x 1 x 512       =   512 halfs

  const int BLK = 256;
  const int gTileIn = ((NT + TMIN - 1) / TMIN + 7) / 8;  // TMIN tiles/wave, 8 waves/block
  const int gTile   = ((NT + TM - 1) / TM + 7) / 8;      // TM tiles/wave, 8 waves/block
  const int gNode   = (NN + BLK - 1) / BLK;
  const int gFeat   = (NN * HH + BLK - 1) / BLK;
  const int gEdgeW  = (ET + 7) / 8;                      // wave per edge, 8 waves/block
  const int gEdgeT  = (ET + BLK - 1) / BLK;

  // pack weights into WMMA fragment layout (runs every launch; deterministic)
  k_pack_w<<<1, BLK, 0, stream>>>(W0, HH, 4, 2, pW0);           // 256 threads
  k_pack_w<<<4, BLK, 0, stream>>>(Ws, HH, 15, 2, pWs);          // 960 threads (kt spans layers)
  k_pack_w<<<1, BLK, 0, stream>>>(W16, DOUT, 1, 1, pW16);       // 32 threads

  k_gemm_in<<<gTileIn, BLK, 0, stream>>>(x, pW0, b0, h);

  for (int l = 0; l < LL; ++l) {
    const _Float16* wp = pWs + l * 2 * 512;
    const float* al = att_l + l * HH;
    const float* ar = att_r + l * HH;
    const float* b  = bs    + l * HH;
    k_gemm_hid<<<gTile, BLK, 0, stream>>>(h, wp, hp);
    k_scores  <<<gNode, BLK, 0, stream>>>(hp, al, ar, sL, sR);
    k_init    <<<gFeat, BLK, 0, stream>>>(amax, den, acc);
    k_edge1   <<<gEdgeW, BLK, 0, stream>>>(ei, hp, sL, sR, alpha, amax);
    k_edge2   <<<gEdgeT, BLK, 0, stream>>>(ei, alpha, amax, den);
    k_edge3   <<<gEdgeW, BLK, 0, stream>>>(ei, hp, alpha, den, acc);
    k_epilogue<<<gFeat, BLK, 0, stream>>>(acc, b, h);
  }

  k_gemm_out<<<gTile, BLK, 0, stream>>>(h, pW16, b16, out);
}
